// EncoderLSTMReal_90477781057687
// MI455X (gfx1250) — matrix operand order repaired and checked by reading, not compiled, not run on hardware
//
#include <hip/hip_runtime.h>
#include <hip/hip_bf16.h>

typedef __attribute__((ext_vector_type(16))) __bf16 v16bf;
typedef __attribute__((ext_vector_type(8)))  float  v8f;
typedef unsigned int u32;

#define T_STEPS 200
#define BATCH   1024
#define XD      48
#define AD      15
#define HID     512
#define OUTD    32
#define INW     64          // XD + AD + 1
#define K4H     2048        // 4*HID
#define KTOT    576         // INW + HID
#define HP1     513         // HID + 1

#define BM 64               // batch rows per block
#define BH 64               // hidden columns per block (x4 gates internally)
#define BK 32               // K chunk
#define LSTR 40             // bf16 elems per LDS row (32 + 8 pad): 80B, 16B-aligned,
                            // 20-word stride -> conflict-free for 16-lane fragment reads

// async 16B global->LDS copy (CDNA5 GLOBAL_LOAD_ASYNC_TO_LDS_B128, ASYNCcnt-tracked)
__device__ __forceinline__ void async_cp16(u32 ldsOff, const void* gptr) {
    asm volatile("global_load_async_to_lds_b128 %0, %1, off"
                 :: "v"(ldsOff), "v"(gptr) : "memory");
}
__device__ __forceinline__ void wait_async0() {
    asm volatile("s_wait_asynccnt 0x0" ::: "memory");
}

// ---------------------------------------------------------------- prologue ---

__global__ void reduce_max_m(const float* __restrict__ m, int n, float* __restrict__ out) {
    __shared__ float sm[256];
    float v = -3.4e38f;
    for (int i = threadIdx.x; i < n; i += 256) v = fmaxf(v, m[i]);
    sm[threadIdx.x] = v;
    __syncthreads();
    for (int s = 128; s > 0; s >>= 1) {
        if ((int)threadIdx.x < s) sm[threadIdx.x] = fmaxf(sm[threadIdx.x], sm[threadIdx.x + s]);
        __syncthreads();
    }
    if (threadIdx.x == 0) *out = sm[0];
}

// WcT[n][k] = (k < INW ? w_ih[n][k] : w_hh[n][k-INW]) as bf16; n in [0,2048), k in [0,576)
__global__ void build_wct(const float* __restrict__ w_ih, const float* __restrict__ w_hh,
                          __bf16* __restrict__ WcT) {
    int idx = blockIdx.x * blockDim.x + threadIdx.x;
    if (idx >= K4H * KTOT) return;
    int n = idx / KTOT, k = idx - n * KTOT;
    float v = (k < INW) ? w_ih[n * INW + k] : w_hh[n * HID + (k - INW)];
    WcT[idx] = (__bf16)v;
}

__global__ void build_bias(const float* __restrict__ b_ih, const float* __restrict__ b_hh,
                           float* __restrict__ bias) {
    int idx = blockIdx.x * blockDim.x + threadIdx.x;
    if (idx < K4H) bias[idx] = b_ih[idx] + b_hh[idx];
}

// Flipped + concatenated bf16 inputs: Xb[t][b][0:48]=x[T-1-t], [48:63]=a[T-1-t], [63]=t/max(m)
__global__ void build_xb(const float* __restrict__ x, const float* __restrict__ a,
                         const float* __restrict__ maxm, __bf16* __restrict__ Xb) {
    int idx = blockIdx.x * blockDim.x + threadIdx.x;
    if (idx >= T_STEPS * BATCH) return;
    int t = idx / BATCH, b = idx - t * BATCH;
    int src = T_STEPS - 1 - t;
    const float* xr = x + (src * BATCH + b) * XD;
    const float* ar = a + (src * BATCH + b) * AD;
    __bf16* dst = Xb + (size_t)idx * INW;
    #pragma unroll
    for (int j = 0; j < XD; ++j) dst[j] = (__bf16)xr[j];
    #pragma unroll
    for (int j = 0; j < AD; ++j) dst[XD + j] = (__bf16)ar[j];
    dst[XD + AD] = (__bf16)((float)t / maxm[0]);
}

__global__ void init_state(__bf16* __restrict__ hbf, float* __restrict__ c) {
    int idx = blockIdx.x * blockDim.x + threadIdx.x;
    if (idx >= BATCH * HID) return;
    hbf[idx] = (__bf16)0.0f;
    c[idx]   = 0.0f;
}

// --------------------------------------------- fused per-step LSTM cell ------
// Each block owns a (64 batch x 64 hidden) cell region and computes all four
// gate column groups (n = g*512 + h) with bf16 WMMA, then applies the LSTM
// nonlinearity in-register. h is ping-ponged (Hin/Hout) to avoid cross-block
// RAW races within a step. LDS tiles are filled with async global->LDS copies
// (ASYNCcnt) double-buffered against the WMMA work.

__device__ __forceinline__ float sigf(float x) { return 1.0f / (1.0f + __expf(-x)); }

__global__ __launch_bounds__(256) void lstm_step(
    const __bf16* __restrict__ Xrow,   // [B][64]   this step's bf16 input rows
    const __bf16* __restrict__ Hin,    // [B][512]  h_{t-1}, bf16
    __bf16*       __restrict__ Hout,   // [B][512]  h_t, bf16
    float*        __restrict__ Cst,    // [B][512]  cell state, f32 (in/out)
    float*        __restrict__ Hf,     // [B][512]  h_t, f32 (for heads; last write wins)
    const __bf16* __restrict__ WcT,    // [2048][576]
    const float*  __restrict__ bias)   // [2048]
{
    __shared__ __bf16 As[2][BM * LSTR];            // 2 x 5 KB
    __shared__ __bf16 Bs[2][4 * BH * LSTR];        // 2 x 20 KB (4 gates)

    const int tid    = threadIdx.x;
    const int lane   = tid & 31;
    const int wave   = tid >> 5;      // 0..7
    const int waveM  = wave >> 2;     // 0..1 -> 32 rows each
    const int waveN  = wave & 3;      // 0..3 -> 16 hidden cols each
    const int bBase  = blockIdx.x * BM;
    const int hBase  = blockIdx.y * BH;
    const int hiHalf = lane >> 4;
    const int fragRow = lane & 15;

    // A-fragment K-pair offsets (ISA 7.12.2, 16-bit A 16x32)
    int kpA[8], kpB[8];
    #pragma unroll
    for (int v = 0; v < 8; ++v) {
        kpA[v] = ((v < 4) ? 2 * v : 16 + 2 * (v - 4)) + (hiHalf ? 8 : 0);
        kpB[v] = ((v < 4) ? 2 * v : 16 + 2 * (v - 4)) + (hiHalf ? 8 : 0);
    }

    // staging indices (16B = 8 bf16 per async copy)
    const int aRow  = tid >> 2;               // 0..63
    const int aQuad = tid & 3;                // 0..3
    v8f acc[2][4] = {};                       // [mi][gate]

    // ---- stage chunk kc into LDS buffer p via async global->LDS b128 copies
    auto stage = [&](int kc, int p) {
        // A: 64 rows x 32 K (one 16B copy per thread)
        const void* ga = (kc < INW)
            ? (const void*)(Xrow + (size_t)(bBase + aRow) * INW + kc + aQuad * 8)
            : (const void*)(Hin + (size_t)(bBase + aRow) * HID + (kc - INW) + aQuad * 8);
        async_cp16((u32)(uintptr_t)(&As[p][aRow * LSTR + aQuad * 8]), ga);
        // B: 4 gates x 64 n-rows x 32 K (four 16B copies per thread)
        #pragma unroll
        for (int i = 0; i < 4; ++i) {
            int s = tid + i * 256;            // 1024 slots
            int g = s >> 8, rr = (s >> 2) & 63, quad = s & 3;
            int n = g * HID + hBase + rr;
            async_cp16((u32)(uintptr_t)(&Bs[p][(g * BH + rr) * LSTR + quad * 8]),
                       (const void*)(WcT + (size_t)n * KTOT + kc + quad * 8));
        }
    };

    stage(0, 0);
    wait_async0();
    __syncthreads();

    int p = 0;
    for (int kc = 0; kc < KTOT; kc += BK) {
        int kn = kc + BK;
        if (kn < KTOT) {
            __builtin_prefetch(WcT + (size_t)(hBase + (tid & 63)) * KTOT + kn + BK, 0, 1);
            stage(kn, p ^ 1);                 // async copies fly during the WMMAs below
        }

        union Frag { v16bf v; u32 u[8]; } Af[2], Bf[4];
        #pragma unroll
        for (int mi = 0; mi < 2; ++mi) {
            int mrow = waveM * 32 + mi * 16 + fragRow;
            #pragma unroll
            for (int v = 0; v < 8; ++v)
                Af[mi].u[v] = *(const u32*)(&As[p][mrow * LSTR + kpA[v]]);
        }
        #pragma unroll
        for (int g = 0; g < 4; ++g) {
            int nrow = g * BH + waveN * 16 + fragRow;
            #pragma unroll
            for (int v = 0; v < 8; ++v)
                Bf[g].u[v] = *(const u32*)(&Bs[p][nrow * LSTR + kpB[v]]);
        }
        #pragma unroll
        for (int mi = 0; mi < 2; ++mi)
            #pragma unroll
            for (int g = 0; g < 4; ++g)
                acc[mi][g] = __builtin_amdgcn_wmma_f32_16x16x32_bf16(
                    false, Af[mi].v, false, Bf[g].v, (short)0, acc[mi][g], false, false);

        wait_async0();                        // my async copies landed...
        __syncthreads();                      // ...and everyone else's too
        p ^= 1;
    }

    // ---- fused LSTM epilogue: C layout lane&15 = N col, VGPR r = row (+8 hi half)
    const int h = hBase + waveN * 16 + fragRow;
    const float bi = bias[0 * HID + h];
    const float bf = bias[1 * HID + h];
    const float bg = bias[2 * HID + h];
    const float bo = bias[3 * HID + h];

    #pragma unroll
    for (int mi = 0; mi < 2; ++mi) {
        int mTile = waveM * 32 + mi * 16;
        #pragma unroll
        for (int r = 0; r < 8; ++r) {
            int b = bBase + mTile + r + (hiHalf ? 8 : 0);
            size_t idx = (size_t)b * HID + h;
            float ig = acc[mi][0][r] + bi;
            float fg = acc[mi][1][r] + bf;
            float gg = acc[mi][2][r] + bg;
            float og = acc[mi][3][r] + bo;
            float cn = sigf(fg) * Cst[idx] + sigf(ig) * tanhf(gg);
            float hn = sigf(og) * tanhf(cn);
            Cst[idx]  = cn;
            Hout[idx] = (__bf16)hn;
            Hf[idx]   = hn;
        }
    }
}

// ------------------------------------------------------------------- heads ---

__global__ void head_l1(const float* __restrict__ hlast, const float* __restrict__ w1,
                        const float* __restrict__ b1, float* __restrict__ z1) {
    int idx = blockIdx.x * blockDim.x + threadIdx.x;      // < B*513
    if (idx >= BATCH * HP1) return;
    int b = idx / HP1, j = idx - b * HP1;
    const float4* hv = (const float4*)(hlast + (size_t)b * HID);
    const float4* wv = (const float4*)(w1 + (size_t)j * HID);
    float s = 0.f;
    #pragma unroll 4
    for (int k = 0; k < HID / 4; ++k) {
        float4 hq = hv[k], wq = wv[k];
        s += hq.x * wq.x + hq.y * wq.y + hq.z * wq.z + hq.w * wq.w;
    }
    z1[idx] = tanhf(s + b1[j]);
}

__global__ void head_l2(const float* __restrict__ z1, const float* __restrict__ w2,
                        const float* __restrict__ b2, float* __restrict__ out) {
    int idx = blockIdx.x * blockDim.x + threadIdx.x;      // < B*32
    if (idx >= BATCH * OUTD) return;
    int b = idx >> 5, o = idx & 31;
    const float* zv = z1 + (size_t)b * HP1;
    const float* wv = w2 + (size_t)o * HP1;
    float s = 0.f;
    for (int k = 0; k < HP1; ++k) s += zv[k] * wv[k];
    out[b * OUTD + o] = tanhf(s + b2[o]);
}

// ------------------------------------------------------------------ launch ---

extern "C" void kernel_launch(void* const* d_in, const int* in_sizes, int n_in,
                              void* d_out, int out_size, void* d_ws, size_t ws_size,
                              hipStream_t stream) {
    const float* x    = (const float*)d_in[0];
    const float* a    = (const float*)d_in[1];
    const float* m    = (const float*)d_in[2];
    const float* w_ih = (const float*)d_in[3];
    const float* w_hh = (const float*)d_in[4];
    const float* b_ih = (const float*)d_in[5];
    const float* b_hh = (const float*)d_in[6];
    const float* lw1  = (const float*)d_in[7];
    const float* lb1  = (const float*)d_in[8];
    const float* lw2  = (const float*)d_in[9];
    const float* lb2  = (const float*)d_in[10];
    const float* vw1  = (const float*)d_in[11];
    const float* vb1  = (const float*)d_in[12];
    const float* vw2  = (const float*)d_in[13];
    const float* vb2  = (const float*)d_in[14];
    float* out = (float*)d_out;

    // workspace carve-up (256B aligned)
    char* w = (char*)d_ws;
    auto alloc = [&](size_t bytes) { char* p = w; w += (bytes + 255) & ~(size_t)255; return p; };
    float*  maxm  = (float*)  alloc(sizeof(float));
    __bf16* WcT   = (__bf16*) alloc((size_t)K4H * KTOT * 2);
    float*  bias  = (float*)  alloc((size_t)K4H * 4);
    __bf16* Xb    = (__bf16*) alloc((size_t)T_STEPS * BATCH * INW * 2);
    __bf16* Hbf0  = (__bf16*) alloc((size_t)BATCH * HID * 2);
    __bf16* Hbf1  = (__bf16*) alloc((size_t)BATCH * HID * 2);
    float*  Cst   = (float*)  alloc((size_t)BATCH * HID * 4);
    float*  Hf    = (float*)  alloc((size_t)BATCH * HID * 4);
    float*  Z1    = (float*)  alloc((size_t)BATCH * HP1 * 4);

    // prologue
    reduce_max_m<<<1, 256, 0, stream>>>(m, T_STEPS * BATCH, maxm);
    build_wct<<<(K4H * KTOT + 255) / 256, 256, 0, stream>>>(w_ih, w_hh, WcT);
    build_bias<<<(K4H + 255) / 256, 256, 0, stream>>>(b_ih, b_hh, bias);
    build_xb<<<(T_STEPS * BATCH + 255) / 256, 256, 0, stream>>>(x, a, maxm, Xb);
    init_state<<<(BATCH * HID + 255) / 256, 256, 0, stream>>>(Hbf0, Cst);

    // fused recurrent sweep: one kernel per step, h ping-ponged
    dim3 ggrid(BATCH / BM, HID / BH);   // 16 x 8 = 128 blocks
    for (int t = 0; t < T_STEPS; ++t) {
        __bf16* Hin  = (t & 1) ? Hbf1 : Hbf0;
        __bf16* Hout = (t & 1) ? Hbf0 : Hbf1;
        lstm_step<<<ggrid, 256, 0, stream>>>(Xb + (size_t)t * BATCH * INW,
                                             Hin, Hout, Cst, Hf, WcT, bias);
    }

    // heads: mu then log_var
    head_l1<<<(BATCH * HP1 + 255) / 256, 256, 0, stream>>>(Hf, lw1, lb1, Z1);
    head_l2<<<(BATCH * OUTD) / 256, 256, 0, stream>>>(Z1, lw2, lb2, out);
    head_l1<<<(BATCH * HP1 + 255) / 256, 256, 0, stream>>>(Hf, vw1, vb1, Z1);
    head_l2<<<(BATCH * OUTD) / 256, 256, 0, stream>>>(Z1, vw2, vb2, out + BATCH * OUTD);
}
